// ConnectorAttention_65859028517561
// MI455X (gfx1250) — matrix operand order repaired and checked by reading, not compiled, not run on hardware
//
#include <hip/hip_runtime.h>
#include <hip/hip_bf16.h>
#include <stdint.h>

#define DIM_C   3840
#define INNER_C 3840
#define HEADS_C 30
#define HD_C    128
#define SEQ_C   2048
#define BATCH_C 2
#define ROWS_C  (BATCH_C * SEQ_C)   // 4096

typedef __attribute__((ext_vector_type(16))) __bf16          bf16x16;
typedef __attribute__((ext_vector_type(8)))  float           f32x8;
typedef __attribute__((ext_vector_type(4)))  float           f32x4;
typedef __attribute__((ext_vector_type(16))) unsigned short  u16x16;
typedef __attribute__((ext_vector_type(8)))  unsigned short  u16x8;
typedef __attribute__((ext_vector_type(4)))  unsigned int    uint32x4;
typedef __attribute__((ext_vector_type(8)))  int             int32x8;
typedef __attribute__((ext_vector_type(4)))  int             int32x4;

#if defined(__gfx1250__) && __has_builtin(__builtin_amdgcn_tensor_load_to_lds)
#define HAVE_TDM 1
#else
#define HAVE_TDM 0
#endif

// fp32 -> bf16 round-to-nearest-even
static __device__ __forceinline__ unsigned short f2bf(float f) {
  unsigned int u = __builtin_bit_cast(unsigned int, f);
  u += 0x7FFFu + ((u >> 16) & 1u);
  return (unsigned short)(u >> 16);
}

// 16x32 bf16 A-fragment (contiguous k = 16h + s): lane holds two 8-element K-chunks
// (lo at kbase, hi at kbase+8).
static __device__ __forceinline__ bf16x16 frag_from_halves(const unsigned short* p0,
                                                           const unsigned short* p1) {
  u16x8 lo = *(const u16x8*)p0;
  u16x8 hi = *(const u16x8*)p1;
  u16x16 c = __builtin_shufflevector(lo, hi, 0,1,2,3,4,5,6,7,8,9,10,11,12,13,14,15);
  return __builtin_bit_cast(bf16x16, c);
}

static __device__ __forceinline__ f32x8 f32x8_zero() {
  f32x8 z = {0.f,0.f,0.f,0.f,0.f,0.f,0.f,0.f};
  return z;
}

// ---------------------------------------------------------------------------
// Issue one TDM 2D tile load: tile_rows rows of tile_d0 contiguous bf16
// elements, source row stride row_stride_elems, into LDS at lds_byte_off.
// D# layout per CDNA5 ISA (2D: groups 2/3 zero).
// Caller must gate to a single wave and pair with s_wait_tensorcnt + barrier.
// ---------------------------------------------------------------------------
static __device__ __forceinline__ void tdm_issue_2d(
    unsigned lds_byte_off, const void* gsrc,
    unsigned tile_d0_elems, unsigned tile_rows,
    unsigned long long row_stride_elems)
{
#if HAVE_TDM
  const unsigned long long ga = (unsigned long long)(size_t)gsrc;
  uint32x4 g0 = {0u, 0u, 0u, 0u};
  g0[0] = 1u;                                           // count=1 (valid user D#)
  g0[1] = lds_byte_off;                                 // lds_addr
  g0[2] = (unsigned)ga;                                 // global_addr[31:0]
  g0[3] = (unsigned)((ga >> 32) & 0x01FFFFFFull) | (2u << 30);  // addr[56:32], type=2
  int32x8 g1 = {0, 0, 0, 0, 0, 0, 0, 0};
  const unsigned td0 = 0x7FFFFFFFu, td1 = 0x7FFFFFFFu;  // huge tensor dims (no OOB clip)
  g1[0] = (int)(1u << 16);                              // wg_mask=0, data_size=1 (2 bytes)
  g1[1] = (int)((td0 & 0xFFFFu) << 16);                 // tensor_dim0[15:0]
  g1[2] = (int)((td0 >> 16) | ((td1 & 0xFFFFu) << 16)); // tensor_dim0[31:16], dim1[15:0]
  g1[3] = (int)((td1 >> 16) | (tile_d0_elems << 16));   // tensor_dim1[31:16], tile_dim0
  g1[4] = (int)(tile_rows & 0xFFFFu);                   // tile_dim1, tile_dim2=0
  g1[5] = (int)(row_stride_elems & 0xFFFFFFFFull);      // tensor_dim0_stride[31:0]
  g1[6] = (int)((row_stride_elems >> 32) & 0xFFFFull);  // tensor_dim0_stride[47:32]
  int32x4 gz = {0, 0, 0, 0};
#if __clang_major__ >= 23
  int32x8 gz8 = {0, 0, 0, 0, 0, 0, 0, 0};
  __builtin_amdgcn_tensor_load_to_lds(g0, g1, gz, gz, gz8, 0);
#else
  __builtin_amdgcn_tensor_load_to_lds(g0, g1, gz, gz, 0);
#endif
#else
  (void)lds_byte_off; (void)gsrc; (void)tile_d0_elems; (void)tile_rows;
  (void)row_stride_elems;
#endif
}

// ---------------------------------------------------------------------------
// Elementwise fp32 -> bf16 (vectorized by 4).
// ---------------------------------------------------------------------------
__global__ __launch_bounds__(256) void convert_f32_to_bf16(
    const float* __restrict__ in, unsigned short* __restrict__ out, long long n4)
{
  const long long i = (long long)blockIdx.x * 256 + threadIdx.x;
  if (i < n4) {
    f32x4 v = ((const f32x4*)in)[i];
    out[i*4+0] = f2bf(v.x); out[i*4+1] = f2bf(v.y);
    out[i*4+2] = f2bf(v.z); out[i*4+3] = f2bf(v.w);
  }
}

// ---------------------------------------------------------------------------
// W[K][N] fp32 -> Wt[N][K] bf16 (32x32 tiles through padded LDS).
// ---------------------------------------------------------------------------
__global__ __launch_bounds__(256) void transpose_convert_bf16(
    const float* __restrict__ W, unsigned short* __restrict__ Wt, int K, int N)
{
  __shared__ float t[32][33];
  const int k0 = blockIdx.y * 32;
  const int n0 = blockIdx.x * 32;
  const int tx = threadIdx.x & 31;
  const int ty = threadIdx.x >> 5;
  #pragma unroll
  for (int i = 0; i < 4; i++) {
    const int r = ty + 8 * i;
    t[r][tx] = W[(size_t)(k0 + r) * N + n0 + tx];
  }
  __syncthreads();
  #pragma unroll
  for (int i = 0; i < 4; i++) {
    const int r = ty + 8 * i;
    Wt[(size_t)(n0 + r) * K + k0 + tx] = f2bf(t[tx][r]);
  }
}

// ---------------------------------------------------------------------------
// GEMM: Out[MxN] = A[MxK](bf16) @ Bt[NxK](bf16, pre-transposed) + bias(fp32).
// Block tile 128x128, BK=64, 8 waves (wave tile 32x64 -> 16 WMMAs/step).
// Double-buffered LDS; tiles staged by the Tensor Data Mover (wave 0 issues,
// s_wait_tensorcnt + barrier synchronize), cooperative-copy fallback.
// ---------------------------------------------------------------------------
template<bool OUT_BF16>
__global__ __launch_bounds__(256) void wmma_gemm_bias(
    const unsigned short* __restrict__ A,
    const unsigned short* __restrict__ Bt,
    const float* __restrict__ bias, void* __restrict__ Outp,
    int M, int N, int K)
{
  __shared__ unsigned short As[2][128 * 64];
  __shared__ unsigned short Bs[2][128 * 64];

  const int tid   = threadIdx.x;
  const int lane  = tid & 31;
  const int wave  = tid >> 5;
  const int waveM = wave & 3;
  const int waveN = wave >> 2;
  const int bm    = blockIdx.y * 128;
  const int bn    = blockIdx.x * 128;
  const int mf    = lane & 15;
  const int khalf = lane >> 4;

  f32x8 acc[2][4];
  #pragma unroll
  for (int i = 0; i < 2; i++)
    #pragma unroll
    for (int j = 0; j < 4; j++) acc[i][j] = f32x8_zero();

  const int nK = K / 64;

#if HAVE_TDM
  if (tid < 32) {
    tdm_issue_2d((unsigned)(size_t)&As[0][0], A  + (size_t)bm * K, 64u, 128u,
                 (unsigned long long)K);
    tdm_issue_2d((unsigned)(size_t)&Bs[0][0], Bt + (size_t)bn * K, 64u, 128u,
                 (unsigned long long)K);
  }
#else
  {
    const int row = tid >> 1;
    const int c0  = (tid & 1) * 32;
    const u16x16* sa = (const u16x16*)(A  + (size_t)(bm + row) * K + c0);
    const u16x16* sb = (const u16x16*)(Bt + (size_t)(bn + row) * K + c0);
    u16x16* da = (u16x16*)&As[0][row * 64 + c0];
    u16x16* db = (u16x16*)&Bs[0][row * 64 + c0];
    da[0] = sa[0]; da[1] = sa[1];
    db[0] = sb[0]; db[1] = sb[1];
  }
#endif

  for (int kt = 0; kt < nK; kt++) {
    const int cur = kt & 1;
    const int nxt = cur ^ 1;
    const int k0n = (kt + 1) * 64;
#if HAVE_TDM
    if (tid < 32) {
      if (kt + 1 < nK) {
        tdm_issue_2d((unsigned)(size_t)&As[nxt][0], A  + (size_t)bm * K + k0n,
                     64u, 128u, (unsigned long long)K);
        tdm_issue_2d((unsigned)(size_t)&Bs[nxt][0], Bt + (size_t)bn * K + k0n,
                     64u, 128u, (unsigned long long)K);
        __builtin_amdgcn_s_wait_tensorcnt(2);   // current tiles complete (in-order)
      } else {
        __builtin_amdgcn_s_wait_tensorcnt(0);
      }
    }
#else
    if (kt + 1 < nK) {
      const int row = tid >> 1;
      const int c0  = (tid & 1) * 32;
      const u16x16* sa = (const u16x16*)(A  + (size_t)(bm + row) * K + k0n + c0);
      const u16x16* sb = (const u16x16*)(Bt + (size_t)(bn + row) * K + k0n + c0);
      u16x16* da = (u16x16*)&As[nxt][row * 64 + c0];
      u16x16* db = (u16x16*)&Bs[nxt][row * 64 + c0];
      da[0] = sa[0]; da[1] = sa[1];
      db[0] = sb[0]; db[1] = sb[1];
    }
#endif
    __syncthreads();

    #pragma unroll
    for (int kc = 0; kc < 2; kc++) {
      bf16x16 aF[2];
      #pragma unroll
      for (int i = 0; i < 2; i++) {
        const unsigned short* rp =
            &As[cur][(waveM * 32 + i * 16 + mf) * 64 + kc * 32 + khalf * 16];
        aF[i] = frag_from_halves(rp, rp + 8);
      }
      #pragma unroll
      for (int j = 0; j < 4; j++) {
        const unsigned short* np =
            &Bs[cur][(waveN * 64 + j * 16 + mf) * 64 + kc * 32 + khalf * 16];
        bf16x16 bF = __builtin_bit_cast(bf16x16, *(const u16x16*)np);
        #pragma unroll
        for (int i = 0; i < 2; i++)
          acc[i][j] = __builtin_amdgcn_wmma_f32_16x16x32_bf16(
              false, aF[i], false, bF, (short)0, acc[i][j], false, false);
      }
    }
    __syncthreads();
  }

  // epilogue: +bias, store (C layout: lane half -> row+8, vgpr r -> row)
  #pragma unroll
  for (int j = 0; j < 4; j++) {
    const int col = bn + waveN * 64 + j * 16 + mf;
    const float bb = bias[col];
    #pragma unroll
    for (int i = 0; i < 2; i++) {
      const int rbase = bm + waveM * 32 + i * 16 + khalf * 8;
      #pragma unroll
      for (int r = 0; r < 8; r++) {
        const float v = acc[i][j][r] + bb;
        if constexpr (OUT_BF16)
          ((unsigned short*)Outp)[(size_t)(rbase + r) * N + col] = f2bf(v);
        else
          ((float*)Outp)[(size_t)(rbase + r) * N + col] = v;
      }
    }
  }
}

// ---------------------------------------------------------------------------
// Row RMSNorm (fp32 in) * gamma -> bf16 out.  One block per row.
// ---------------------------------------------------------------------------
__global__ __launch_bounds__(256) void rmsnorm_to_bf16(
    const float* __restrict__ X, const float* __restrict__ G,
    unsigned short* __restrict__ Out, int C)
{
  const int row = blockIdx.x;
  const float* x = X + (size_t)row * C;
  float ss = 0.f;
  for (int c = threadIdx.x; c < C; c += 256) { const float v = x[c]; ss += v * v; }
  #pragma unroll
  for (int off = 16; off > 0; off >>= 1) ss += __shfl_xor(ss, off, 32);
  __shared__ float red[8];
  if ((threadIdx.x & 31) == 0) red[threadIdx.x >> 5] = ss;
  __syncthreads();
  float tot = 0.f;
  #pragma unroll
  for (int i = 0; i < 8; i++) tot += red[i];
  const float scale = rsqrtf(tot / (float)C + 1e-6f);
  for (int c = threadIdx.x; c < C; c += 256)
    Out[(size_t)row * C + c] = f2bf(x[c] * G[c] * scale);
}

// ---------------------------------------------------------------------------
// Flash attention (non-causal), bf16 in/out, fp32 accumulate.
// Block = 128 q-rows x one head x one batch; 8 waves, 16 q-rows/wave.
// Q tile + K chunks staged by TDM; V transposed cooperatively.
// ---------------------------------------------------------------------------
__global__ __launch_bounds__(256) void flash_attn_wmma(
    const unsigned short* __restrict__ Qb,
    const unsigned short* __restrict__ Kb,
    const unsigned short* __restrict__ Vb,
    unsigned short* __restrict__ Ob)
{
  __shared__ unsigned short Qs[128 * 128];  // [q][d]
  __shared__ unsigned short Ks[64 * 128];   // [key][d]  (B [n][k] for QK^T)
  __shared__ unsigned short Vst[128 * 64];  // [d][key]  (B [n][k] for P@V)
  __shared__ unsigned short Ps[128 * 64];   // [q][key] bf16 probabilities

  const int tid   = threadIdx.x;
  const int lane  = tid & 31;
  const int wave  = tid >> 5;
  const int mf    = lane & 15;
  const int khalf = lane >> 4;

  const int qt = blockIdx.x, h = blockIdx.y, b = blockIdx.z;
  const size_t stride  = INNER_C;
  const int    colBase = h * HD_C;
  const size_t qRow0   = (size_t)b * SEQ_C + (size_t)qt * 128;
  const size_t kRow0   = (size_t)b * SEQ_C;

#if HAVE_TDM
  if (tid < 32)
    tdm_issue_2d((unsigned)(size_t)&Qs[0], Qb + qRow0 * stride + colBase,
                 128u, 128u, (unsigned long long)stride);
#else
  {
    const int r  = tid >> 1;
    const int c0 = (tid & 1) * 64;
    const unsigned short* src = Qb + (qRow0 + r) * stride + colBase + c0;
    u16x16* dst = (u16x16*)&Qs[r * 128 + c0];
    #pragma unroll
    for (int i = 0; i < 4; i++) dst[i] = ((const u16x16*)src)[i];
  }
#endif

  f32x8 o[8];
  #pragma unroll
  for (int n = 0; n < 8; n++) o[n] = f32x8_zero();
  float m_run[8], l_run[8];
  #pragma unroll
  for (int r = 0; r < 8; r++) { m_run[r] = -1e30f; l_run[r] = 0.f; }

  const int   waveRow = wave * 16;
  const float scale   = 0.088388347648318447f;  // 1/sqrt(128)

  for (int kc = 0; kc < SEQ_C; kc += 64) {
    __syncthreads();  // previous iteration done reading Ks/Vst
#if HAVE_TDM
    if (tid < 32)
      tdm_issue_2d((unsigned)(size_t)&Ks[0], Kb + (kRow0 + kc) * stride + colBase,
                   128u, 64u, (unsigned long long)stride);
#else
    {
      const int r  = tid >> 2;
      const int c0 = (tid & 3) * 32;
      const unsigned short* src = Kb + (kRow0 + kc + r) * stride + colBase + c0;
      u16x16* dst = (u16x16*)&Ks[r * 128 + c0];
      dst[0] = ((const u16x16*)src)[0];
      dst[1] = ((const u16x16*)src)[1];
    }
#endif
    // stage V chunk transposed: Vst[d][key] (overlaps with TDM)
    {
      const int r  = tid >> 2;
      const int c0 = (tid & 3) * 32;
      const unsigned short* src = Vb + (kRow0 + kc + r) * stride + colBase + c0;
      u16x16 v0 = ((const u16x16*)src)[0];
      u16x16 v1 = ((const u16x16*)src)[1];
      #pragma unroll
      for (int i = 0; i < 16; i++) Vst[(c0 + i) * 64 + r] = v0[i];
      #pragma unroll
      for (int i = 0; i < 16; i++) Vst[(c0 + 16 + i) * 64 + r] = v1[i];
    }
#if HAVE_TDM
    if (tid < 32) __builtin_amdgcn_s_wait_tensorcnt(0);
#endif
    __syncthreads();

    // S = Q @ K^T : 16 rows x 64 keys, K-depth 128 in 4 steps
    f32x8 s[4];
    #pragma unroll
    for (int j = 0; j < 4; j++) s[j] = f32x8_zero();
    #pragma unroll
    for (int ks = 0; ks < 4; ks++) {
      const unsigned short* ap = &Qs[(waveRow + mf) * 128 + ks * 32 + khalf * 16];
      bf16x16 aF = frag_from_halves(ap, ap + 8);
      #pragma unroll
      for (int j = 0; j < 4; j++) {
        const unsigned short* bp = &Ks[(j * 16 + mf) * 128 + ks * 32 + khalf * 16];
        bf16x16 bF = __builtin_bit_cast(bf16x16, *(const u16x16*)bp);
        s[j] = __builtin_amdgcn_wmma_f32_16x16x32_bf16(
            false, aF, false, bF, (short)0, s[j], false, false);
      }
    }

    // online softmax (row of this lane group = khalf*8 + r; width-16 shuffles)
    #pragma unroll
    for (int r = 0; r < 8; r++) {
      float mx = s[0][r];
      mx = fmaxf(mx, s[1][r]); mx = fmaxf(mx, s[2][r]); mx = fmaxf(mx, s[3][r]);
      mx *= scale;
      #pragma unroll
      for (int off = 8; off > 0; off >>= 1) mx = fmaxf(mx, __shfl_xor(mx, off, 16));
      const float mnew  = fmaxf(m_run[r], mx);
      const float alpha = __expf(m_run[r] - mnew);
      float lsum = 0.f;
      #pragma unroll
      for (int j = 0; j < 4; j++) {
        const float p = __expf(s[j][r] * scale - mnew);
        lsum += p;
        Ps[(waveRow + khalf * 8 + r) * 64 + j * 16 + mf] = f2bf(p);
      }
      #pragma unroll
      for (int off = 8; off > 0; off >>= 1) lsum += __shfl_xor(lsum, off, 16);
      l_run[r] = l_run[r] * alpha + lsum;
      m_run[r] = mnew;
      #pragma unroll
      for (int n = 0; n < 8; n++) o[n][r] *= alpha;
    }

    // O += P @ V (wave reads only its own Ps rows; same-wave LDS is in-order)
    #pragma unroll
    for (int ks = 0; ks < 2; ks++) {
      const unsigned short* ap = &Ps[(waveRow + mf) * 64 + ks * 32 + khalf * 16];
      bf16x16 aF = frag_from_halves(ap, ap + 8);
      #pragma unroll
      for (int n = 0; n < 8; n++) {
        const unsigned short* bp = &Vst[(n * 16 + mf) * 64 + ks * 32 + khalf * 16];
        bf16x16 bF = __builtin_bit_cast(bf16x16, *(const u16x16*)bp);
        o[n] = __builtin_amdgcn_wmma_f32_16x16x32_bf16(
            false, aF, false, bF, (short)0, o[n], false, false);
      }
    }
  }

  // finalize: divide by l, store bf16 head-slice
  #pragma unroll
  for (int n = 0; n < 8; n++) {
    const int col = colBase + n * 16 + mf;
    #pragma unroll
    for (int r = 0; r < 8; r++) {
      const size_t row = qRow0 + waveRow + khalf * 8 + r;
      Ob[row * stride + col] = f2bf(o[n][r] / l_run[r]);
    }
  }
}

// ---------------------------------------------------------------------------
extern "C" void kernel_launch(void* const* d_in, const int* in_sizes, int n_in,
                              void* d_out, int out_size, void* d_ws, size_t ws_size,
                              hipStream_t stream)
{
  (void)in_sizes; (void)n_in; (void)out_size; (void)ws_size;
  const float* x  = (const float*)d_in[0];
  const float* Wq = (const float*)d_in[1];
  const float* bq = (const float*)d_in[2];
  const float* Wk = (const float*)d_in[3];
  const float* bk = (const float*)d_in[4];
  const float* Wv = (const float*)d_in[5];
  const float* bv = (const float*)d_in[6];
  const float* Wo = (const float*)d_in[7];
  const float* bo = (const float*)d_in[8];
  const float* gq = (const float*)d_in[9];
  const float* gk = (const float*)d_in[10];

  size_t off = 0;
  auto alloc = [&](size_t bytes) -> void* {
    void* p = (char*)d_ws + off;
    off += (bytes + 255) & ~(size_t)255;
    return p;
  };
  unsigned short* xb  = (unsigned short*)alloc((size_t)ROWS_C * DIM_C * 2);
  unsigned short* Wqt = (unsigned short*)alloc((size_t)DIM_C * INNER_C * 2);
  unsigned short* Wkt = (unsigned short*)alloc((size_t)DIM_C * INNER_C * 2);
  unsigned short* Wvt = (unsigned short*)alloc((size_t)DIM_C * INNER_C * 2);
  unsigned short* Wot = (unsigned short*)alloc((size_t)INNER_C * DIM_C * 2);
  float*          qf  = (float*)alloc((size_t)ROWS_C * INNER_C * 4);
  float*          kf  = (float*)alloc((size_t)ROWS_C * INNER_C * 4);
  unsigned short* vbf = (unsigned short*)alloc((size_t)ROWS_C * INNER_C * 2);
  unsigned short* qbf = (unsigned short*)alloc((size_t)ROWS_C * INNER_C * 2);
  unsigned short* kbf = (unsigned short*)alloc((size_t)ROWS_C * INNER_C * 2);
  unsigned short* abf = (unsigned short*)alloc((size_t)ROWS_C * INNER_C * 2);

  const dim3 blk(256);
  const long long xN4 = (long long)ROWS_C * DIM_C / 4;
  convert_f32_to_bf16<<<dim3((unsigned)((xN4 + 255) / 256)), blk, 0, stream>>>(x, xb, xN4);

  const dim3 gT(INNER_C / 32, DIM_C / 32);   // (120, 120)
  transpose_convert_bf16<<<gT, blk, 0, stream>>>(Wq, Wqt, DIM_C, INNER_C);
  transpose_convert_bf16<<<gT, blk, 0, stream>>>(Wk, Wkt, DIM_C, INNER_C);
  transpose_convert_bf16<<<gT, blk, 0, stream>>>(Wv, Wvt, DIM_C, INNER_C);
  transpose_convert_bf16<<<gT, blk, 0, stream>>>(Wo, Wot, INNER_C, DIM_C);

  const dim3 gQKV(INNER_C / 128, ROWS_C / 128);   // (30, 32)
  wmma_gemm_bias<false><<<gQKV, blk, 0, stream>>>(xb, Wqt, bq, qf,  ROWS_C, INNER_C, DIM_C);
  wmma_gemm_bias<false><<<gQKV, blk, 0, stream>>>(xb, Wkt, bk, kf,  ROWS_C, INNER_C, DIM_C);
  wmma_gemm_bias<true ><<<gQKV, blk, 0, stream>>>(xb, Wvt, bv, vbf, ROWS_C, INNER_C, DIM_C);

  rmsnorm_to_bf16<<<ROWS_C, blk, 0, stream>>>(qf, gq, qbf, INNER_C);
  rmsnorm_to_bf16<<<ROWS_C, blk, 0, stream>>>(kf, gk, kbf, INNER_C);

  flash_attn_wmma<<<dim3(SEQ_C / 128, HEADS_C, BATCH_C), blk, 0, stream>>>(qbf, kbf, vbf, abf);

  wmma_gemm_bias<false><<<dim3(DIM_C / 128, ROWS_C / 128), blk, 0, stream>>>(
      abf, Wot, bo, d_out, ROWS_C, DIM_C, INNER_C);
}
